// MixCrossAttention_88046829568162
// MI455X (gfx1250) — matrix operand order repaired and checked by reading, not compiled, not run on hardware
//
#include <hip/hip_runtime.h>
#include <hip/hip_bf16.h>
#include <math.h>

// Problem constants (from reference)
#define BB 16
#define LL 512
#define SS 512
#define HH 8
#define EE 64
#define MM 64   // MODES
// IN_CH = OUT_CH = 512, scale = 1/(512*512)

typedef __attribute__((ext_vector_type(16))) _Float16 v16h;
typedef __attribute__((ext_vector_type(8)))  _Float16 v8h;
typedef __attribute__((ext_vector_type(8)))  float    v8f;

#define WMMA_F16(a,b,c) __builtin_amdgcn_wmma_f32_16x16x32_f16(false,(a),false,(b),(short)0,(c),false,false)

// ---------------------------------------------------------------------------
// WMMA fragment loaders (wave32 layouts per CDNA5 ISA 7.12.2), vectorized:
// every operand = two 16-byte LDS loads (lane data is contiguous by layout).
//
// A 16x32 f16 row-major [m][k]: lane (m = lane&15, sect = lane>>4) needs
//   halves k0+sect*8 .. +7  (vgpr 0-3)  and  k0+16+sect*8 .. +7 (vgpr 4-7)
// B 32x16 f16 stored [n][k]: lane (n = n0+(lane&15), sect) needs halves
//   k0+sect*8 .. +7  and  k0+16+sect*8 .. +7
// ---------------------------------------------------------------------------
__device__ __forceinline__ v16h load_a_vec(const _Float16* A, int lda,
                                           int m0, int k0, int lane) {
  int row  = m0 + (lane & 15);
  int sect = lane >> 4;
  const _Float16* p = A + row * lda + k0 + sect * 8;
  v8h lo = *(const v8h*)p;
  v8h hi = *(const v8h*)(p + 16);
  return __builtin_shufflevector(lo, hi, 0, 1, 2, 3, 4, 5, 6, 7,
                                 8, 9, 10, 11, 12, 13, 14, 15);
}

__device__ __forceinline__ v16h load_b_vec(const _Float16* Bm, int ldb,
                                           int k0, int n0, int lane,
                                           int kmask) {
  int n    = n0 + (lane & 15);
  int sect = lane >> 4;
  int ks   = (k0 + sect * 8) & kmask;   // runs of 8 (+16) never cross fold boundary
  const _Float16* p = Bm + n * ldb + ks;
  v8h lo = *(const v8h*)p;
  v8h hi = *(const v8h*)(p + 16);
  return __builtin_shufflevector(lo, hi, 0, 1, 2, 3, 4, 5, 6, 7,
                                 8, 9, 10, 11, 12, 13, 14, 15);
}

// ---------------------------------------------------------------------------
// Phase B: circular autocorrelation.  grid = B*H*E, block = 256.
// corr_t layout [B,H,L,E]
// ---------------------------------------------------------------------------
__global__ __launch_bounds__(256)
void autocorr_kernel(const float* __restrict__ q, float* __restrict__ corr_t) {
  __shared__ float xs[LL];
  __shared__ float red[256];
  __shared__ float rr[LL];
  int blk = blockIdx.x;          // (b*8+h)*64 + e
  int e   = blk & 63;
  int bh  = blk >> 6;
  int h   = bh & 7;
  int b   = bh >> 3;
  int tid = threadIdx.x;

  for (int t = tid; t < LL; t += 256)
    xs[t] = q[(((size_t)b * LL + t) * HH + h) * EE + e];
  __syncthreads();
  red[tid] = xs[tid] + xs[tid + 256];
  __syncthreads();
  for (int off = 128; off > 0; off >>= 1) {
    if (tid < off) red[tid] += red[tid + off];
    __syncthreads();
  }
  float mean = red[0] * (1.0f / LL);
  __syncthreads();
  for (int t = tid; t < LL; t += 256) xs[t] -= mean;
  __syncthreads();

  for (int lag = tid; lag < LL; lag += 256) {
    float acc = 0.f;
    for (int t = 0; t < LL; ++t)
      acc += xs[t] * xs[(t + lag) & (LL - 1)];
    rr[lag] = acc;
  }
  __syncthreads();
  float r0inv = 1.0f / rr[0];
  for (int lag = tid; lag < LL; lag += 256)
    corr_t[((size_t)bh * LL + lag) * EE + e] = rr[lag] * r0inv;
}

// ---------------------------------------------------------------------------
// Phase D1: sparse DFT at selected modes.  grid = B*H*E, block = 64 (=MODES)
// out layout [B,H,E,M]
// ---------------------------------------------------------------------------
__global__ __launch_bounds__(64)
void dft_kernel(const float* __restrict__ x, const int* __restrict__ idx,
                float* __restrict__ ore, float* __restrict__ oim) {
  __shared__ float xs[LL];
  int blk = blockIdx.x;
  int e   = blk & 63;
  int bh  = blk >> 6;
  int h   = bh & 7;
  int b   = bh >> 3;
  int tid = threadIdx.x;

  for (int t = tid; t < LL; t += 64)
    xs[t] = x[(((size_t)b * LL + t) * HH + h) * EE + e];
  __syncthreads();

  int k = idx[tid];
  const float C = 6.2831853071795864769f / (float)LL;
  float re = 0.f, im = 0.f;
  for (int t = 0; t < LL; ++t) {
    int p = (k * t) & (LL - 1);
    float th = (float)p * C;
    float s, c;
    __sincosf(th, &s, &c);
    re += xs[t] * c;
    im -= xs[t] * s;
  }
  ore[(size_t)blk * MM + tid] = re;
  oim[(size_t)blk * MM + tid] = im;
}

// ---------------------------------------------------------------------------
// Phase D2+D3: mode attention, per (b,h).  grid = B*H, block = 256,
// dynamic LDS = 6 * 64*64 * 4 = 98304 bytes.
// xqkv layout [B,H,E,M]
// ---------------------------------------------------------------------------
__global__ __launch_bounds__(256)
void mode_attn_kernel(const float* __restrict__ qre, const float* __restrict__ qim,
                      const float* __restrict__ kre, const float* __restrict__ kim,
                      float* __restrict__ vre_out, float* __restrict__ vim_out) {
  extern __shared__ float sm[];
  float* Qr = sm;               // [E][M]
  float* Qi = Qr + 4096;
  float* Kr = Qi + 4096;
  float* Ki = Kr + 4096;
  float* Xr = Ki + 4096;        // xqk [M][M]
  float* Xi = Xr + 4096;

  int bh  = blockIdx.x;
  int tid = threadIdx.x;
  size_t base = (size_t)bh * 4096;
  for (int i = tid; i < 4096; i += 256) {
    Qr[i] = qre[base + i]; Qi[i] = qim[base + i];
    Kr[i] = kre[base + i]; Ki[i] = kim[base + i];
  }
  __syncthreads();

  // xqk[x][y] = tanh_c( sum_e q[e][x]*k[e][y] )
  for (int j = 0; j < 16; ++j) {
    int o = tid * 16 + j;
    int x = o >> 6, y = o & 63;
    float ar = 0.f, ai = 0.f;
    for (int e = 0; e < EE; ++e) {
      float qr = Qr[e * 64 + x], qi = Qi[e * 64 + x];
      float kr = Kr[e * 64 + y], ki = Ki[e * 64 + y];
      ar += qr * kr - qi * ki;
      ai += qr * ki + qi * kr;
    }
    float t2a = 2.f * ar, t2b = 2.f * ai;
    float tre, tim;
    if (fabsf(t2a) > 20.f) {
      tre = (t2a > 0.f) ? 1.f : -1.f;
      tim = 0.f;
    } else {
      float d = coshf(t2a) + cosf(t2b);
      tre = sinhf(t2a) / d;
      tim = sinf(t2b) / d;
    }
    Xr[x * 64 + y] = tre;
    Xi[x * 64 + y] = tim;
  }
  __syncthreads();

  // xqkv[e][x] = sum_y xqk[x][y]*k[e][y]
  for (int j = 0; j < 16; ++j) {
    int o = tid * 16 + j;
    int e = o >> 6, x = o & 63;
    float ar = 0.f, ai = 0.f;
    for (int y = 0; y < MM; ++y) {
      float xr = Xr[x * 64 + y], xi = Xi[x * 64 + y];
      float kr = Kr[e * 64 + y], ki = Ki[e * 64 + y];
      ar += xr * kr - xi * ki;
      ai += xr * ki + xi * kr;
    }
    vre_out[base + e * 64 + x] = ar;
    vim_out[base + e * 64 + x] = ai;
  }
}

// ---------------------------------------------------------------------------
// Phase D4: xqkvw[b,h,o,x] = sum_e xqkv[b,h,e,x]*w[h,e,o,x] (complex)
// grid = H*M (h = blk>>6, x = blk&63), block = 256
// ---------------------------------------------------------------------------
__global__ __launch_bounds__(256)
void mode_weight_kernel(const float* __restrict__ vre, const float* __restrict__ vim,
                        const float* __restrict__ wre, const float* __restrict__ wim,
                        float* __restrict__ ore, float* __restrict__ oim) {
  __shared__ float Wr[4096], Wi[4096];   // [e][o]
  __shared__ float Vr[1024], Vi[1024];   // [b][e]
  int blk = blockIdx.x;
  int h = blk >> 6, x = blk & 63;
  int tid = threadIdx.x;
  for (int i = tid; i < 4096; i += 256) {
    int e = i >> 6, o = i & 63;
    size_t a = (((size_t)h * EE + e) * EE + o) * MM + x;
    Wr[i] = wre[a]; Wi[i] = wim[a];
  }
  for (int i = tid; i < 1024; i += 256) {
    int b = i >> 6, e = i & 63;
    size_t a = (((size_t)b * HH + h) * EE + e) * MM + x;
    Vr[i] = vre[a]; Vi[i] = vim[a];
  }
  __syncthreads();
  for (int j = 0; j < 4; ++j) {
    int oi = tid * 4 + j;
    int b = oi >> 6, o = oi & 63;
    float ar = 0.f, ai = 0.f;
    for (int e = 0; e < EE; ++e) {
      float xr = Vr[b * 64 + e], xi = Vi[b * 64 + e];
      float yr = Wr[e * 64 + o], yi = Wi[e * 64 + o];
      ar += xr * yr - xi * yi;
      ai += xr * yi + xi * yr;
    }
    size_t a = (((size_t)b * HH + h) * EE + o) * MM + x;
    ore[a] = ar; oim[a] = ai;
  }
}

// ---------------------------------------------------------------------------
// Phase D5: sparse inverse rFFT.  grid = B*H*E (per output channel), block=256
// freq layout [B,H,E,L]; scale = 1/(L * IN_CH * OUT_CH)
// ---------------------------------------------------------------------------
__global__ __launch_bounds__(256)
void idft_kernel(const float* __restrict__ wre, const float* __restrict__ wim,
                 const int* __restrict__ idx, float* __restrict__ freq) {
  __shared__ float Rr[MM], Ri[MM];
  __shared__ int   Kk[MM];
  __shared__ float Cm[MM];
  int blk = blockIdx.x;
  int tid = threadIdx.x;
  if (tid < MM) {
    Rr[tid] = wre[(size_t)blk * MM + tid];
    Ri[tid] = wim[(size_t)blk * MM + tid];
    int k = idx[tid];
    Kk[tid] = k;
    Cm[tid] = (k == 0) ? 1.f : 2.f;
  }
  __syncthreads();
  const float C   = 6.2831853071795864769f / (float)LL;
  const float inv = 1.0f / ((float)LL * 262144.0f);
  for (int l = tid; l < LL; l += 256) {
    float acc = 0.f;
    for (int m = 0; m < MM; ++m) {
      int p = (Kk[m] * l) & (LL - 1);
      float th = (float)p * C;
      float s, c;
      __sincosf(th, &s, &c);
      acc += Cm[m] * (Rr[m] * c - Ri[m] * s);
    }
    freq[(size_t)blk * LL + l] = acc * inv;
  }
}

// ---------------------------------------------------------------------------
// Phase C: fused mixed-scores GEMM + softmax + A@V GEMM (WMMA).
// grid = B*H*(L/64), block = 512 (16 wave32).
// dynamic LDS layout (bytes):
//   Ah     [64][128] f16          @ 0        (16384)
//   Bh     xk [s][72] / val [e][520] f16
//                                 @ 16384    (73728)
//   mixS   [64][516]  f32         @ 90112    (132096)
//   probsH [64][528]  f16         @ 222208   (67584)
//   rowInv [64]       f32         @ 289792   (256)
//   total 290048  (<= 320KB WGP LDS)
// ---------------------------------------------------------------------------
__global__ __launch_bounds__(512)
void attn_fused_kernel(const float* __restrict__ tfq_src,  // tf_queries [B,L,H,E]
                       const float* __restrict__ keys,     // [B,S,H,E]
                       const float* __restrict__ values,   // [B,S,H,E]
                       const float* __restrict__ tw,
                       const float* __restrict__ fw,
                       const float* __restrict__ corr_t,   // [B,H,L,E]
                       const float* __restrict__ freq,     // [B,H,E,L]
                       float* __restrict__ out) {          // [B,H,E,L]
  extern __shared__ char smem[];
  _Float16* Ah     = (_Float16*)smem;                   // 64 x 128
  _Float16* Bh     = (_Float16*)(smem + 16384);
  float*    mixS   = (float*)(smem + 90112);            // 64 x 516
  _Float16* probsH = (_Float16*)(smem + 222208);        // 64 x 528
  float*    rowInv = (float*)(smem + 289792);
  const int MSTR = 516;
  const int PSTR = 528;

  int blk  = blockIdx.x;
  int bh   = blk >> 3;       // b*8+h
  int lt   = blk & 7;        // L tile
  int b    = bh >> 3, h = bh & 7;
  int tid  = threadIdx.x;
  int lane = tid & 31, wv = tid >> 5;

  // ---- fill A = [ wt[l]*corr(l,e) | wf[l]*tfq(l,e) ], 64 x 128 ----
  {
    int lrow = tid >> 3;
    int kb   = (tid & 7) * 16;
    int lg   = lt * 64 + lrow;
    float twv = tw[lg], fwv = fw[lg];
    float tot = twv + fwv;
    if (tot == 0.f) tot = 1e-8f;
    float wtv = twv / tot, wfv = fwv / tot;
    const float* crow = corr_t + ((size_t)bh * LL + lg) * EE;
    const float* trow = tfq_src + (((size_t)b * LL + lg) * HH + h) * EE;
#pragma unroll
    for (int j = 0; j < 16; ++j) {
      int k = kb + j;
      float v = (k < 64) ? wtv * crow[k] : wfv * trow[k - 64];
      Ah[lrow * 128 + k] = (_Float16)v;
    }
  }
  // ---- stage xk as [s][e] (B layout [n][k]), stride 72, vector stores ----
  {
    int s = tid;
    const float* krow = keys + (((size_t)b * SS + s) * HH + h) * EE;
    _Float16 tmp[64];
#pragma unroll
    for (int e = 0; e < EE; ++e) tmp[e] = (_Float16)krow[e];
    v8h* dst = (v8h*)(Bh + s * 72);
#pragma unroll
    for (int i = 0; i < 8; ++i) dst[i] = *(const v8h*)(tmp + i * 8);
  }
  __syncthreads();

  // ---- GEMM1: mix[64 x 512], K = 128 with B-row fold (k & 63) ----
  {
    int mt    = wv & 3;
    int nbase = wv >> 2;
    for (int jj = 0; jj < 8; ++jj) {
      int nt = nbase + 4 * jj;
      v8f c = {};
#pragma unroll
      for (int kc = 0; kc < 4; ++kc) {
        v16h a  = load_a_vec(Ah, 128, mt * 16, kc * 32, lane);
        v16h bb = load_b_vec(Bh, 72, kc * 32, nt * 16, lane, 63);
        c = WMMA_F16(a, bb, c);
      }
      int n    = nt * 16 + (lane & 15);
      int sect = lane >> 4;
#pragma unroll
      for (int r = 0; r < 8; ++r) {
        int m = mt * 16 + r + 8 * sect;
        mixS[m * MSTR + n] = c[r];
      }
    }
  }
  __syncthreads();

  // ---- row softmax over s (scale = 1/2^18 applied first) ----
  if (tid < 64) {
    const float SC = 1.0f / 262144.0f;
    float mx = -3.4e38f;
    for (int s2 = 0; s2 < SS; ++s2) {
      float v = mixS[tid * MSTR + s2] * SC;
      mixS[tid * MSTR + s2] = v;
      mx = fmaxf(mx, v);
    }
    float sum = 0.f;
    for (int s2 = 0; s2 < SS; ++s2) {
      float e2 = __expf(mixS[tid * MSTR + s2] - mx);
      mixS[tid * MSTR + s2] = e2;
      sum += e2;
    }
    rowInv[tid] = 1.0f / sum;
  }
  __syncthreads();

  // ---- convert probs to f16 (fold 1/sum), and stage values as [e][s] ----
  {
    int m   = tid >> 3;
    int kc0 = (tid & 7) * 64;
    float ri = rowInv[m];
#pragma unroll 8
    for (int j = 0; j < 64; ++j)
      probsH[m * PSTR + kc0 + j] = (_Float16)(mixS[m * MSTR + kc0 + j] * ri);
  }
  {
    int e  = tid >> 3;
    int sc = (tid & 7) * 64;
    _Float16 tmp[64];
#pragma unroll
    for (int i = 0; i < 64; ++i)
      tmp[i] = (_Float16)values[(((size_t)b * SS + sc + i) * HH + h) * EE + e];
    v8h* dst = (v8h*)(Bh + e * 520 + sc);
#pragma unroll
    for (int i = 0; i < 8; ++i) dst[i] = *(const v8h*)(tmp + i * 8);
  }
  __syncthreads();

  // ---- GEMM2: V[64 l x 64 e] = softmax(mix) @ values, K = 512 ----
  {
    int mt = wv & 3, nt = wv >> 2;
    v8f c = {};
#pragma unroll
    for (int kc = 0; kc < 16; ++kc) {
      v16h a  = load_a_vec(probsH, PSTR, mt * 16, kc * 32, lane);
      v16h bb = load_b_vec(Bh, 520, kc * 32, nt * 16, lane, 0x7fffffff);
      c = WMMA_F16(a, bb, c);
    }
    int n    = nt * 16 + (lane & 15);
    int sect = lane >> 4;
#pragma unroll
    for (int r = 0; r < 8; ++r) {
      int m  = mt * 16 + r + 8 * sect;
      int lg = lt * 64 + m;
      size_t idx = ((size_t)bh * EE + n) * LL + lg;
      out[idx] = 0.5f * (c[r] + freq[idx]);
    }
  }
}

// ---------------------------------------------------------------------------
// Host launcher
// ---------------------------------------------------------------------------
extern "C" void kernel_launch(void* const* d_in, const int* in_sizes, int n_in,
                              void* d_out, int out_size, void* d_ws, size_t ws_size,
                              hipStream_t stream) {
  const float* tfq  = (const float*)d_in[0];
  const float* qry  = (const float*)d_in[1];
  const float* keys = (const float*)d_in[2];
  const float* vals = (const float*)d_in[3];
  // d_in[4] = mask (unused by reference)
  const float* tw   = (const float*)d_in[5];
  const float* fw   = (const float*)d_in[6];
  const float* w1re = (const float*)d_in[7];
  const float* w1im = (const float*)d_in[8];
  const int*   idxq = (const int*)d_in[9];
  const int*   idxk = (const int*)d_in[10];
  float* out = (float*)d_out;
  float* ws  = (float*)d_ws;

  const size_t NBHE_L = (size_t)BB * HH * EE * LL;   // 4194304
  const size_t NBHE_M = (size_t)BB * HH * EE * MM;   // 524288

  float* corr    = ws;
  float* qftre   = corr    + NBHE_L;
  float* qftim   = qftre   + NBHE_M;
  float* kftre   = qftim   + NBHE_M;
  float* kftim   = kftre   + NBHE_M;
  float* xqkvre  = kftim   + NBHE_M;
  float* xqkvim  = xqkvre  + NBHE_M;
  float* xqkvwre = xqkvim  + NBHE_M;
  float* xqkvwim = xqkvwre + NBHE_M;
  float* freq    = xqkvwim + NBHE_M;

  // Phase B: autocorrelation -> corr [B,H,L,E]
  autocorr_kernel<<<BB * HH * EE, 256, 0, stream>>>(qry, corr);
  // Phase D1: sparse DFTs
  dft_kernel<<<BB * HH * EE, 64, 0, stream>>>(qry, idxq, qftre, qftim);
  dft_kernel<<<BB * HH * EE, 64, 0, stream>>>(keys, idxk, kftre, kftim);
  // Phase D2/3: complex-tanh mode attention
  mode_attn_kernel<<<BB * HH, 256, 98304, stream>>>(qftre, qftim, kftre, kftim,
                                                    xqkvre, xqkvim);
  // Phase D4: per-mode complex weight contraction
  mode_weight_kernel<<<HH * MM, 256, 0, stream>>>(xqkvre, xqkvim, w1re, w1im,
                                                  xqkvwre, xqkvwim);
  // Phase D5: sparse inverse rFFT -> freq [B,H,E,L]
  idft_kernel<<<BB * HH * EE, 256, 0, stream>>>(xqkvwre, xqkvwim, idxq, freq);
  // Phase C: fused WMMA attention + (V + freq)/2
  const size_t ldsC = 290048;
  attn_fused_kernel<<<BB * HH * (LL / 64), 512, ldsC, stream>>>(
      tfq, keys, vals, tw, fw, corr, freq, out);
}
